// NeuroSAT_15006615732429
// MI455X (gfx1250) — matrix operand order repaired and checked by reading, not compiled, not run on hardware
//
#include <hip/hip_runtime.h>
#include <cstddef>

// ---------------------------------------------------------------------------
// NeuroSAT on MI455X (gfx1250): bf16 WMMA GEMM pipeline.
//  - M (4000x8000 fp32, binary values) converted once per launch to bf16
//    in both orientations (exact, halves HBM traffic on the dominant GEMMs).
//  - All weights transposed to [Kpad, Npad] bf16 with zero padding so every
//    matmul maps onto v_wmma_f32_16x16x32_bf16 (f32 accumulate).
//  - LSTM cell state stays f32; hidden states stored bf16 (WMMA A operand).
//  - GEMM: double-buffered LDS B-staging (1 barrier / K-step), next B chunk
//    register-prefetched during WMMA, A prefetched via global_prefetch.
// ---------------------------------------------------------------------------

typedef __attribute__((ext_vector_type(16))) __bf16 v16bf;
typedef __attribute__((ext_vector_type(8)))  float v8f;

union Frag16 {
    uint4 q[2];
    unsigned short h[16];
    v16bf v;
};

__device__ __forceinline__ unsigned short f32_to_bf16(float f) {
    unsigned int u = __float_as_uint(f);
    unsigned int r = u + 0x7FFFu + ((u >> 16) & 1u);   // round to nearest even
    return (unsigned short)(r >> 16);
}
__device__ __forceinline__ float bf16_to_f32(unsigned short h) {
    return __uint_as_float(((unsigned int)h) << 16);
}
__device__ __forceinline__ float sigf(float x) { return 1.0f / (1.0f + expf(-x)); }

#define TM 64
#define TN 128

// C[M,N] = A[M,K](bf16,row-major) x B[K,N](bf16,row-major) (+bias)(+Cf32)(relu?)
// M multiple of 16, K multiple of 32 (guaranteed by padding); N edge-guarded.
template <bool RELU, bool ACCUM, bool HAS_BIAS, bool WF32, bool WBF16>
__global__ __launch_bounds__(256)
void gemm_bf16_wmma(const unsigned short* __restrict__ A, int lda,
                    const unsigned short* __restrict__ B, int ldb,
                    const float* __restrict__ bias,
                    float* __restrict__ Cf, unsigned short* __restrict__ Cb,
                    int ldc, int Mdim, int Ndim, int Kdim)
{
    // B tile staged in fragment-swizzled order: halves index
    //   ((s*32 + (k>>4)*16 + (col&15)) << 4) + (k&15)
    // so each lane reads its v16bf B fragment as two contiguous b128 LDS loads.
    // Double buffered: write buf[cur] -> barrier -> prefetch next -> WMMA.
    __shared__ unsigned short Bs[2][32 * TN];

    const int t    = threadIdx.x;
    const int lane = t & 31;
    const int w    = t >> 5;       // 0..7
    const int wm   = w & 3;        // 4 row-subtiles of 16
    const int wn   = w >> 2;       // 2 col-groups of 64
    const int rowBase = blockIdx.y * TM + wm * 16;
    const int colTile = blockIdx.x * TN;
    const bool active = rowBase < Mdim;           // wave-uniform (EXEC stays full)
    const int aRow = rowBase + (lane & 15);
    const int kgA  = lane >> 4;

    v8f acc[4];
#pragma unroll
    for (int j = 0; j < 4; ++j)
        acc[j] = (v8f){0.f, 0.f, 0.f, 0.f, 0.f, 0.f, 0.f, 0.f};

    const int sr   = t >> 4;              // staging k row (chunk adds +16)
    const int sc   = (t & 15) * 8;        // staging col start
    const int scol = colTile + sc;
    const bool sin = scol < Ndim;         // N multiple of 16, sc multiple of 8

    // prefetch first B chunk into registers
    uint4 d[2];
#pragma unroll
    for (int ch = 0; ch < 2; ++ch) {
        uint4 z = {0u, 0u, 0u, 0u};
        d[ch] = z;
        if (sin)
            d[ch] = *reinterpret_cast<const uint4*>(B + (size_t)(sr + ch * 16) * ldb + scol);
    }

    int cur = 0;
    for (int k0 = 0; k0 < Kdim; k0 += 32) {
        // write the staged registers into LDS buf[cur] (fragment-swizzled)
#pragma unroll
        for (int ch = 0; ch < 2; ++ch) {
            const int r = sr + ch * 16;            // k within 32-step
            const unsigned short* hh = reinterpret_cast<const unsigned short*>(&d[ch]);
            const int lgk = r >> 4;                // B layout: k = lg*16 + e
            const int e   = r & 15;
#pragma unroll
            for (int j = 0; j < 8; ++j) {
                const int c = sc + j;
                Bs[cur][((((c >> 4) * 32) + lgk * 16 + (c & 15)) << 4) + e] = hh[j];
            }
        }
        __syncthreads();

        // issue next chunk's global loads now; latency overlaps the WMMAs below
        const bool more = (k0 + 32) < Kdim;
        if (more) {
#pragma unroll
            for (int ch = 0; ch < 2; ++ch) {
                uint4 z = {0u, 0u, 0u, 0u};
                d[ch] = z;
                if (sin)
                    d[ch] = *reinterpret_cast<const uint4*>(
                        B + (size_t)(k0 + 32 + sr + ch * 16) * ldb + scol);
            }
        }

        if (active) {
            const unsigned short* ap = A + (size_t)aRow * lda + k0 + kgA * 16;
            if (more)
                __builtin_prefetch(ap + 32, 0, 1);   // next K-step A fragment
            Frag16 af;
            af.q[0] = *reinterpret_cast<const uint4*>(ap);        // K = kg*16..+7
            af.q[1] = *reinterpret_cast<const uint4*>(ap + 8);    // K = kg*16+8..+15
#pragma unroll
            for (int j = 0; j < 4; ++j) {
                const int s = wn * 4 + j;
                Frag16 bf;
                const unsigned short* bp = &Bs[cur][(s * 32 + lane) << 4];
                bf.q[0] = *reinterpret_cast<const uint4*>(bp);
                bf.q[1] = *reinterpret_cast<const uint4*>(bp + 8);
                acc[j] = __builtin_amdgcn_wmma_f32_16x16x32_bf16(
                    false, af.v, false, bf.v, (short)0, acc[j], false, false);
            }
        }
        cur ^= 1;
    }

    if (!active) return;
    const int rb = rowBase + ((lane >> 4) << 3);
#pragma unroll
    for (int j = 0; j < 4; ++j) {
        const int col = colTile + (wn * 4 + j) * 16 + (lane & 15);
        if (col >= Ndim) continue;
        const float bv = HAS_BIAS ? bias[col] : 0.0f;
#pragma unroll
        for (int r = 0; r < 8; ++r) {
            const size_t idx = (size_t)(rb + r) * ldc + col;
            float v = acc[j][r] + bv;
            if (ACCUM) v += Cf[idx];
            if (RELU)  v = v > 0.0f ? v : 0.0f;
            if (WF32)  Cf[idx] = v;
            if (WBF16) Cb[idx] = f32_to_bf16(v);
        }
    }
}

// M (fp32 [R,C]) -> Mb bf16 [R,C] and MTb bf16 [C,R]   (R,C multiples of 32)
__global__ __launch_bounds__(256)
void conv_transpose_M(const float* __restrict__ M,
                      unsigned short* __restrict__ Mb,
                      unsigned short* __restrict__ MTb, int R, int C)
{
    __shared__ float tile[32][33];
    const int tc = blockIdx.x * 32, tr = blockIdx.y * 32;
    const int c = threadIdx.x, r0 = threadIdx.y;     // 32 x 8
#pragma unroll
    for (int i = 0; i < 4; ++i) {
        const int r = r0 + i * 8;
        const float v = M[(size_t)(tr + r) * C + tc + c];
        tile[r][c] = v;
        Mb[(size_t)(tr + r) * C + tc + c] = f32_to_bf16(v);
    }
    __syncthreads();
#pragma unroll
    for (int i = 0; i < 4; ++i) {
        const int r = r0 + i * 8;
        MTb[(size_t)(tc + r) * R + tr + c] = f32_to_bf16(tile[c][r]);
    }
}

// W fp32 [outD,inD] -> Wt bf16 [Kpad,Npad], Wt[k][n] = W[n][k] (zero padded)
__global__ void pack_weight(const float* __restrict__ W, unsigned short* __restrict__ Wt,
                            int outD, int inD, int Kpad, int Npad)
{
    const int idx = blockIdx.x * blockDim.x + threadIdx.x;
    if (idx >= Kpad * Npad) return;
    const int k = idx / Npad, n = idx % Npad;
    const float v = (k < inD && n < outD) ? W[(size_t)n * inD + k] : 0.0f;
    Wt[idx] = f32_to_bf16(v);
}

__global__ void pack_bias(const float* __restrict__ b, float* __restrict__ bp, int n, int npad)
{
    const int idx = blockIdx.x * blockDim.x + threadIdx.x;
    if (idx >= npad) return;
    bp[idx] = (idx < n) ? b[idx] : 0.0f;
}

__global__ void init_state(const float* __restrict__ init, unsigned short* __restrict__ hb,
                           float* __restrict__ c, int nrows)
{
    const int idx = blockIdx.x * blockDim.x + threadIdx.x;
    if ((idx >> 7) >= nrows) return;
    hb[idx] = f32_to_bf16(init[idx & 127]);
    c[idx]  = 0.0f;
}

__global__ void lstm_update(const float* __restrict__ gates, float* __restrict__ c,
                            unsigned short* __restrict__ hb, int nrows)
{
    const int idx = blockIdx.x * blockDim.x + threadIdx.x;
    const int n = idx >> 7, e = idx & 127;
    if (n >= nrows) return;
    const float* g = gates + (size_t)n * 512;
    const float i  = sigf(g[e]);
    const float f  = sigf(g[128 + e]);
    const float gg = tanhf(g[256 + e]);
    const float o  = sigf(g[384 + e]);
    const float c2 = f * c[idx] + i * gg;
    c[idx]  = c2;
    hb[idx] = f32_to_bf16(o * tanhf(c2));
}

__global__ __launch_bounds__(256)
void vote_partial(const unsigned short* __restrict__ H2, const float* __restrict__ W3,
                  const float* __restrict__ b3, float* __restrict__ partial,
                  int nrows, int ld)
{
    __shared__ float red[256];
    const int n = blockIdx.x * 256 + threadIdx.x;
    float v = 0.0f;
    if (n < nrows) {
        const unsigned short* h = H2 + (size_t)n * ld;
        float acc = 0.0f;
        for (int k = 0; k < 200; ++k) acc += bf16_to_f32(h[k]) * W3[k];
        v = sigf(acc + b3[0]);
    }
    red[threadIdx.x] = v;
    __syncthreads();
    for (int s = 128; s > 0; s >>= 1) {
        if (threadIdx.x < s) red[threadIdx.x] += red[threadIdx.x + s];
        __syncthreads();
    }
    if (threadIdx.x == 0) partial[blockIdx.x] = red[0];
}

__global__ void finalize_logit(const float* __restrict__ partial, int nblocks,
                               float* __restrict__ out, float denom)
{
    if (threadIdx.x == 0) {
        float s = 0.0f;
        for (int i = 0; i < nblocks; ++i) s += partial[i];
        const float avg = s / denom;
        out[0] = logf(avg / (1.0f - avg));
    }
}

// ---------------------------------------------------------------------------

extern "C" void kernel_launch(void* const* d_in, const int* in_sizes, int n_in,
                              void* d_out, int out_size, void* d_ws, size_t ws_size,
                              hipStream_t stream)
{
    (void)in_sizes; (void)n_in; (void)out_size; (void)ws_size;

    const int NL = 4000, NC = 8000, E = 128;
    const int H1P = 416, H2P = 224, G = 512;   // padded MLP dims (K x32, N x16)
    const int T = 30;

    const float* M_f    = (const float*)d_in[0];
    const float* L_init = (const float*)d_in[1];
    const float* C_init = (const float*)d_in[2];
    const float* LC_W1 = (const float*)d_in[3];  const float* LC_b1 = (const float*)d_in[4];
    const float* LC_W2 = (const float*)d_in[5];  const float* LC_b2 = (const float*)d_in[6];
    const float* LC_W3 = (const float*)d_in[7];  const float* LC_b3 = (const float*)d_in[8];
    const float* CL_W1 = (const float*)d_in[9];  const float* CL_b1 = (const float*)d_in[10];
    const float* CL_W2 = (const float*)d_in[11]; const float* CL_b2 = (const float*)d_in[12];
    const float* CL_W3 = (const float*)d_in[13]; const float* CL_b3 = (const float*)d_in[14];
    const float* L_Wih = (const float*)d_in[15]; const float* L_Whh = (const float*)d_in[16];
    const float* L_b   = (const float*)d_in[17];
    const float* C_Wih = (const float*)d_in[18]; const float* C_Whh = (const float*)d_in[19];
    const float* C_b   = (const float*)d_in[20];
    const float* V_W1 = (const float*)d_in[21];  const float* V_b1 = (const float*)d_in[22];
    const float* V_W2 = (const float*)d_in[23];  const float* V_b2 = (const float*)d_in[24];
    const float* V_W3 = (const float*)d_in[25];  const float* V_b3 = (const float*)d_in[26];

    // ---- workspace bump allocator -----------------------------------------
    size_t off = 0;
    auto alloc = [&](size_t bytes) -> void* {
        void* p = (char*)d_ws + off;
        off += (bytes + 255) & ~(size_t)255;
        return p;
    };
    unsigned short* Mb   = (unsigned short*)alloc((size_t)NL * NC * 2);
    unsigned short* MTb  = (unsigned short*)alloc((size_t)NL * NC * 2);
    unsigned short* lcw1 = (unsigned short*)alloc((size_t)E * H1P * 2);
    unsigned short* lcw2 = (unsigned short*)alloc((size_t)H1P * H2P * 2);
    unsigned short* lcw3 = (unsigned short*)alloc((size_t)H2P * E * 2);
    unsigned short* clw1 = (unsigned short*)alloc((size_t)E * H1P * 2);
    unsigned short* clw2 = (unsigned short*)alloc((size_t)H1P * H2P * 2);
    unsigned short* clw3 = (unsigned short*)alloc((size_t)H2P * E * 2);
    unsigned short* vw1  = (unsigned short*)alloc((size_t)E * H1P * 2);
    unsigned short* vw2  = (unsigned short*)alloc((size_t)H1P * H2P * 2);
    unsigned short* lwih = (unsigned short*)alloc((size_t)E * G * 2);
    unsigned short* lwhh = (unsigned short*)alloc((size_t)E * G * 2);
    unsigned short* cwih = (unsigned short*)alloc((size_t)E * G * 2);
    unsigned short* cwhh = (unsigned short*)alloc((size_t)E * G * 2);
    float* lcb1 = (float*)alloc(H1P * 4);
    float* lcb2 = (float*)alloc(H2P * 4);
    float* clb1 = (float*)alloc(H1P * 4);
    float* clb2 = (float*)alloc(H2P * 4);
    float* vb1  = (float*)alloc(H1P * 4);
    float* vb2  = (float*)alloc(H2P * 4);
    unsigned short* Lhb = (unsigned short*)alloc((size_t)NL * E * 2);
    unsigned short* Chb = (unsigned short*)alloc((size_t)NC * E * 2);
    float* Lc = (float*)alloc((size_t)NL * E * 4);
    float* Cc = (float*)alloc((size_t)NC * E * 4);
    unsigned short* H1buf  = (unsigned short*)alloc((size_t)NC * H1P * 2);
    unsigned short* H2buf  = (unsigned short*)alloc((size_t)NC * H2P * 2);
    unsigned short* mlpOut = (unsigned short*)alloc((size_t)NC * E * 2);
    unsigned short* xb     = (unsigned short*)alloc((size_t)NC * E * 2);
    float* gatesC = (float*)alloc((size_t)NC * G * 4);
    float* gatesL = (float*)alloc((size_t)NL * G * 4);
    float* partial = (float*)alloc(64 * 4);

    // ---- one-time (per launch) data preparation ---------------------------
    conv_transpose_M<<<dim3(NC / 32, NL / 32), dim3(32, 8), 0, stream>>>(M_f, Mb, MTb, NL, NC);

    struct WDesc { const float* src; unsigned short* dst; int outD, inD, Kp, Np; };
    const WDesc wd[] = {
        {LC_W1, lcw1, 400, 128, 128, H1P}, {LC_W2, lcw2, 200, 400, H1P, H2P}, {LC_W3, lcw3, 128, 200, H2P, 128},
        {CL_W1, clw1, 400, 128, 128, H1P}, {CL_W2, clw2, 200, 400, H1P, H2P}, {CL_W3, clw3, 128, 200, H2P, 128},
        {V_W1,  vw1,  400, 128, 128, H1P}, {V_W2,  vw2,  200, 400, H1P, H2P},
        {L_Wih, lwih, 512, 128, 128, G},   {L_Whh, lwhh, 512, 128, 128, G},
        {C_Wih, cwih, 512, 128, 128, G},   {C_Whh, cwhh, 512, 128, 128, G},
    };
    for (const WDesc& d : wd) {
        const int n = d.Kp * d.Np;
        pack_weight<<<(n + 255) / 256, 256, 0, stream>>>(d.src, d.dst, d.outD, d.inD, d.Kp, d.Np);
    }
    pack_bias<<<(H1P + 255) / 256, 256, 0, stream>>>(LC_b1, lcb1, 400, H1P);
    pack_bias<<<(H2P + 255) / 256, 256, 0, stream>>>(LC_b2, lcb2, 200, H2P);
    pack_bias<<<(H1P + 255) / 256, 256, 0, stream>>>(CL_b1, clb1, 400, H1P);
    pack_bias<<<(H2P + 255) / 256, 256, 0, stream>>>(CL_b2, clb2, 200, H2P);
    pack_bias<<<(H1P + 255) / 256, 256, 0, stream>>>(V_b1, vb1, 400, H1P);
    pack_bias<<<(H2P + 255) / 256, 256, 0, stream>>>(V_b2, vb2, 200, H2P);

    init_state<<<(NL * E) / 256, 256, 0, stream>>>(L_init, Lhb, Lc, NL);
    init_state<<<(NC * E) / 256, 256, 0, stream>>>(C_init, Chb, Cc, NC);

    // ---- GEMM launch helper ----------------------------------------------
    #define GEMM(RELU, ACC, BIAS, WF, WB, A, lda, B, ldb, bias, Cf, Cb, ldc, Md, Nd, Kd)            \
        gemm_bf16_wmma<RELU, ACC, BIAS, WF, WB>                                                     \
            <<<dim3(((Nd) + TN - 1) / TN, ((Md) + TM - 1) / TM), 256, 0, stream>>>(                 \
                A, lda, B, ldb, bias, Cf, Cb, ldc, Md, Nd, Kd)

    // ---- message-passing iterations --------------------------------------
    for (int step = 0; step < T; ++step) {
        // literal MLP: Lhb -> mlpOut (bf16 [NL,128])
        GEMM(true,  false, true,  false, true, Lhb,   E,   lcw1, H1P, lcb1,  nullptr, H1buf,  H1P, NL, H1P, E);
        GEMM(true,  false, true,  false, true, H1buf, H1P, lcw2, H2P, lcb2,  nullptr, H2buf,  H2P, NL, H2P, H1P);
        GEMM(false, false, true,  false, true, H2buf, H2P, lcw3, E,   LC_b3, nullptr, mlpOut, E,   NL, E,   H2P);
        // xC = M^T @ mlpOut   -> xb (bf16 [NC,128])
        GEMM(false, false, false, false, true, MTb,   NL,  mlpOut, E, nullptr, nullptr, xb,   E,   NC, E,   NL);
        // gatesC = xC@Wih^T + b ; gatesC += Ch@Whh^T   (f32 [NC,512])
        GEMM(false, false, true,  true, false, xb,    E,   cwih, G,   C_b,   gatesC,  nullptr, G,  NC, G,   E);
        GEMM(false, true,  false, true, false, Chb,   E,   cwhh, G,   nullptr, gatesC, nullptr, G, NC, G,   E);
        // clause MLP on PRE-update Ch: Chb -> mlpOut (bf16 [NC,128])
        GEMM(true,  false, true,  false, true, Chb,   E,   clw1, H1P, clb1,  nullptr, H1buf,  H1P, NC, H1P, E);
        GEMM(true,  false, true,  false, true, H1buf, H1P, clw2, H2P, clb2,  nullptr, H2buf,  H2P, NC, H2P, H1P);
        GEMM(false, false, true,  false, true, H2buf, H2P, clw3, E,   CL_b3, nullptr, mlpOut, E,   NC, E,   H2P);
        // xL = M @ mlpOut   -> xb (bf16 [NL,128])
        GEMM(false, false, false, false, true, Mb,    NC,  mlpOut, E, nullptr, nullptr, xb,   E,   NL, E,   NC);
        // gatesL = xL@Wih^T + b ; gatesL += Lh@Whh^T   (f32 [NL,512])
        GEMM(false, false, true,  true, false, xb,    E,   lwih, G,   L_b,   gatesL,  nullptr, G,  NL, G,   E);
        GEMM(false, true,  false, true, false, Lhb,   E,   lwhh, G,   nullptr, gatesL, nullptr, G, NL, G,   E);
        // pointwise LSTM updates (writes new Chb/Cc, Lhb/Lc)
        lstm_update<<<(NC * E + 255) / 256, 256, 0, stream>>>(gatesC, Cc, Chb, NC);
        lstm_update<<<(NL * E + 255) / 256, 256, 0, stream>>>(gatesL, Lc, Lhb, NL);
    }

    // ---- vote head + logit ------------------------------------------------
    GEMM(true, false, true, false, true, Lhb,   E,   vw1, H1P, vb1, nullptr, H1buf, H1P, NL, H1P, E);
    GEMM(true, false, true, false, true, H1buf, H1P, vw2, H2P, vb2, nullptr, H2buf, H2P, NL, H2P, H1P);
    const int vblocks = (NL + 255) / 256;   // 16
    vote_partial<<<vblocks, 256, 0, stream>>>(H2buf, V_W3, V_b3, partial, NL, H2P);
    finalize_logit<<<1, 32, 0, stream>>>(partial, vblocks, (float*)d_out, (float)NL);

    #undef GEMM
}
